// GAT_23364622090638
// MI455X (gfx1250) — matrix operand order repaired and checked by reading, not compiled, not run on hardware
//
#include <hip/hip_runtime.h>
#include <hip/hip_bf16.h>
#include <stdint.h>

// Problem constants (from reference)
#define N_NODES   5000
#define IN_F      64
#define HID_F     32
#define N_HEADS   4
#define OUT_F     64
#define NEG_SLOPE 0.2f

// Layout constants
#define N_WORDS    157   // ceil(5000/32) bitmask words per row
#define BITS_PITCH 160   // padded dwords per bitmask row
#define WHT_PITCH  5024  // halves per transposed-feature row (mult of 16 -> 32B aligned tiles)

typedef _Float16 v16h __attribute__((ext_vector_type(16)));
typedef float    v8f  __attribute__((ext_vector_type(8)));

union AFrag { v16h v; _Float16 e[16]; };

__device__ __forceinline__ float lrelu(float e) { return fmaxf(e, NEG_SLOPE * e); }

// ---------------------------------------------------------------------------
// 1) Pack dense fp32 adjacency into a per-row bitmask (one 100MB read, ever).
// ---------------------------------------------------------------------------
__global__ void gat_pack_adj(const float* __restrict__ adj, uint32_t* __restrict__ bits) {
  const int row  = blockIdx.x;
  const int w    = blockIdx.y * 8 + (threadIdx.x >> 5);
  const int lane = threadIdx.x & 31;
  if (w >= N_WORDS) return;
  const int j = w * 32 + lane;
  const bool nb = (j < N_NODES) && (adj[(size_t)row * N_NODES + j] > 0.0f);
#if __has_builtin(__builtin_amdgcn_ballot_w32)
  uint32_t m = __builtin_amdgcn_ballot_w32(nb);
#else
  uint32_t m = (uint32_t)__ballot(nb);
#endif
  if (lane == 0) bits[(size_t)row * BITS_PITCH + w] = m;
}

// ---------------------------------------------------------------------------
// 2) Small feature GEMM Y = X @ W in fp32; also emit f16 transposed copy
//    (feature-major) so WMMA B tiles are contiguous 32B loads.
// ---------------------------------------------------------------------------
template<int K, int C>
__global__ void gat_feat(const float* __restrict__ X, const float* __restrict__ W,
                         float* __restrict__ Yf32, _Float16* __restrict__ YT) {
  const int idx  = blockIdx.x * blockDim.x + threadIdx.x;
  const int node = idx / C;
  const int c    = idx % C;
  if (node >= N_NODES) return;
  const float* xr = X + (size_t)node * K;
  float acc = 0.0f;
#pragma unroll 4
  for (int f = 0; f < K; ++f) acc = fmaf(xr[f], W[f * C + c], acc);
  Yf32[(size_t)node * C + c]       = acc;
  YT[(size_t)c * WHT_PITCH + node] = (_Float16)acc;
}

// ---------------------------------------------------------------------------
// 3) Attention score vectors: s_src[n,h], s_dst[n,h]
// ---------------------------------------------------------------------------
template<int HEADS, int F>
__global__ void gat_scores(const float* __restrict__ Wh, const float* __restrict__ aSrc,
                           const float* __restrict__ aDst,
                           float* __restrict__ sSrc, float* __restrict__ sDst) {
  const int idx  = blockIdx.x * blockDim.x + threadIdx.x;
  const int node = idx / HEADS;
  const int hh   = idx % HEADS;
  if (node >= N_NODES) return;
  const float* wr = Wh + (size_t)node * (HEADS * F) + (size_t)hh * F;
  float s0 = 0.0f, s1 = 0.0f;
#pragma unroll
  for (int f = 0; f < F; ++f) {
    s0 = fmaf(wr[f], aSrc[hh * F + f], s0);
    s1 = fmaf(wr[f], aDst[hh * F + f], s1);
  }
  sSrc[(size_t)node * HEADS + hh] = s0;
  sDst[(size_t)node * HEADS + hh] = s1;
}

// ---------------------------------------------------------------------------
// 4) Per-row softmax stats (max m, sum l) over neighbors, online, one wave/row.
//    Bitmask is L2-resident (3.1MB); s_dst is 80KB.
// ---------------------------------------------------------------------------
template<int HEADS>
__global__ void gat_softmax_stats(const uint32_t* __restrict__ bits,
                                  const float* __restrict__ sSrc,
                                  const float* __restrict__ sDst,
                                  float* __restrict__ mOut, float* __restrict__ lOut) {
  const int row  = blockIdx.x * 8 + (threadIdx.x >> 5);
  const int lane = threadIdx.x & 31;
  if (row >= N_NODES) return;
  float sr[HEADS], m[HEADS], l[HEADS];
#pragma unroll
  for (int h = 0; h < HEADS; ++h) {
    sr[h] = sSrc[(size_t)row * HEADS + h];
    m[h]  = -3.0e38f;
    l[h]  = 0.0f;
  }
  for (int w = lane; w < N_WORDS; w += 32) {
    uint32_t bw = bits[(size_t)row * BITS_PITCH + w];
    while (bw) {
      const int b = __ffs(bw) - 1;
      bw &= bw - 1;
      const int j = w * 32 + b;
#pragma unroll
      for (int h = 0; h < HEADS; ++h) {
        const float e  = lrelu(sr[h] + sDst[(size_t)j * HEADS + h]);
        const float mn = fmaxf(m[h], e);
        l[h] = l[h] * __expf(m[h] - mn) + __expf(e - mn);
        m[h] = mn;
      }
    }
  }
  // wave32 tree combine of (m,l)
#pragma unroll
  for (int off = 16; off >= 1; off >>= 1) {
#pragma unroll
    for (int h = 0; h < HEADS; ++h) {
      const float mo = __shfl_xor(m[h], off, 32);
      const float lo = __shfl_xor(l[h], off, 32);
      const float mn = fmaxf(m[h], mo);
      l[h] = l[h] * __expf(m[h] - mn) + lo * __expf(mo - mn);
      m[h] = mn;
    }
  }
  if (lane == 0) {
#pragma unroll
    for (int h = 0; h < HEADS; ++h) {
      mOut[(size_t)row * HEADS + h] = m[h];
      lOut[(size_t)row * HEADS + h] = l[h];
    }
  }
}

// ---------------------------------------------------------------------------
// 5) WMMA aggregation: out[i, h*FPH + f] = (1/l_i) * sum_j exp(e_ij - m_i) Wh[j,h,f]
//    One wave per 16-row tile. Per 32-j tile: issue ALL B-tile loads first,
//    then build the unnormalized-probability A fragments with pure arithmetic
//    (mask folded in as a 0/1 float multiply -> no divergent branches, EXEC
//    stays all-ones for WMMA), then accumulate with v_wmma_f32_16x16x32_f16.
//    The 1/l normalization is applied once to the f32 accumulators at the end.
// ---------------------------------------------------------------------------
template<int HEADS, int FPH, bool DOELU>
__global__ void __launch_bounds__(32)
gat_aggregate(const uint32_t* __restrict__ bits, const _Float16* __restrict__ whT,
              const float* __restrict__ sSrc, const float* __restrict__ sDst,
              const float* __restrict__ mArr, const float* __restrict__ lArr,
              float* __restrict__ out) {
  constexpr int NT = FPH / 16;
  const int lane = threadIdx.x;   // 0..31 (wave32)
  const int r    = lane & 15;
  const int hi   = lane >> 4;
  const int i0   = blockIdx.x * 16;
  const int rowR = i0 + r;
  const int rowC = rowR < N_NODES ? rowR : (N_NODES - 1);

  float sr[HEADS], mv[HEADS], rl[HEADS];
#pragma unroll
  for (int h = 0; h < HEADS; ++h) {
    sr[h] = sSrc[(size_t)rowC * HEADS + h];
    mv[h] = mArr[(size_t)rowC * HEADS + h];
    const float lv = lArr[(size_t)rowC * HEADS + h];
    rl[h] = 1.0f / (lv > 0.0f ? lv : 1.0f);
  }

  __shared__ float sdl[HEADS * 32];   // s_dst tile, per-head contiguous in j

  v8f acc[HEADS][NT] = {};

  for (int t = 0; t < N_WORDS; ++t) {
    const int j0 = t * 32;
    {
      const int   j  = j0 + lane;
      const int   jc = j < N_NODES ? j : (N_NODES - 1);
      const float zf = (j < N_NODES) ? 1.0f : 0.0f;
#pragma unroll
      for (int h = 0; h < HEADS; ++h)
        sdl[h * 32 + lane] = sDst[(size_t)jc * HEADS + h] * zf;
    }
    __syncthreads();

    const uint32_t bw = bits[(size_t)rowC * BITS_PITCH + t];
    if (t + 1 < N_WORDS)
      __builtin_prefetch(&bits[(size_t)rowC * BITS_PITCH + t + 1], 0, 3);

    // Issue all B-tile loads up front (aligned 32B each) so they overlap
    // with the exp-heavy A-fragment construction below.
    v16h btile[HEADS][NT];
#pragma unroll
    for (int h = 0; h < HEADS; ++h)
#pragma unroll
      for (int tt = 0; tt < NT; ++tt) {
        const _Float16* bp =
            whT + (size_t)(h * FPH + tt * 16 + r) * WHT_PITCH + j0 + hi * 16;
        btile[h][tt] = *(const v16h*)bp;
      }

#pragma unroll
    for (int h = 0; h < HEADS; ++h) {
      // A fragment: 16x32 f16, contiguous layout: VGPR v holds K = 2v,
      // lanes 16..31 add +16; lane holds row M = lane&15.
      AFrag a;
#pragma unroll
      for (int v = 0; v < 8; ++v) {
        const int kb = 2 * v + hi * 16;
        const float2 sd = *(const float2*)&sdl[h * 32 + kb];   // ds_load_b64
        // Unconditional arithmetic; adjacency bit becomes a 0/1 multiplier.
        const float msk0 = (float)((bw >> kb)       & 1u);
        const float msk1 = (float)((bw >> (kb + 1)) & 1u);
        const float p0 = __expf(lrelu(sr[h] + sd.x) - mv[h]) * msk0;
        const float p1 = __expf(lrelu(sr[h] + sd.y) - mv[h]) * msk1;
        a.e[2 * v]     = (_Float16)p0;
        a.e[2 * v + 1] = (_Float16)p1;
      }
#pragma unroll
      for (int tt = 0; tt < NT; ++tt) {
        acc[h][tt] = __builtin_amdgcn_wmma_f32_16x16x32_f16(
            false, a.v, false, btile[h][tt], (short)0, acc[h][tt], false, false);
      }
    }
    __syncthreads();
  }

  // C/D layout: lanes 0-15 -> M=k, lanes 16-31 -> M=k+8; N = lane&15.
  // 1/l for accumulator row (k + hi*8) lives in lane (k + hi*8).
#pragma unroll
  for (int h = 0; h < HEADS; ++h) {
#pragma unroll
    for (int k = 0; k < 8; ++k) {
      const float rls  = __shfl(rl[h], k + hi * 8, 32);
      const int   orow = i0 + k + hi * 8;
      if (orow < N_NODES) {
#pragma unroll
        for (int tt = 0; tt < NT; ++tt) {
          float v = acc[h][tt][k] * rls;
          if (DOELU) v = v > 0.0f ? v : (__expf(v) - 1.0f);
          out[(size_t)orow * (HEADS * FPH) + h * FPH + tt * 16 + r] = v;
        }
      }
    }
  }
}

// ---------------------------------------------------------------------------
extern "C" void kernel_launch(void* const* d_in, const int* in_sizes, int n_in,
                              void* d_out, int out_size, void* d_ws, size_t ws_size,
                              hipStream_t stream) {
  const float* adj = (const float*)d_in[0];
  const float* hIn = (const float*)d_in[1];
  const float* W1  = (const float*)d_in[2];
  const float* a1s = (const float*)d_in[3];
  const float* a1d = (const float*)d_in[4];
  const float* W2  = (const float*)d_in[5];
  const float* a2s = (const float*)d_in[6];
  const float* a2d = (const float*)d_in[7];
  float* out = (float*)d_out;
  (void)in_sizes; (void)n_in; (void)out_size; (void)ws_size;

  uint8_t* ws = (uint8_t*)d_ws;
  size_t off = 0;
  auto carve = [&](size_t bytes) -> void* {
    void* p = ws + off;
    off = (off + bytes + 255) & ~(size_t)255;
    return p;
  };
  uint32_t* bits = (uint32_t*)carve((size_t)N_NODES * BITS_PITCH * 4);
  float*    wh1  = (float*)   carve((size_t)N_NODES * (N_HEADS * HID_F) * 4);
  _Float16* wh1T = (_Float16*)carve((size_t)(N_HEADS * HID_F) * WHT_PITCH * 2);
  float*    s1s  = (float*)   carve((size_t)N_NODES * N_HEADS * 4);
  float*    s1d  = (float*)   carve((size_t)N_NODES * N_HEADS * 4);
  float*    m1   = (float*)   carve((size_t)N_NODES * N_HEADS * 4);
  float*    l1   = (float*)   carve((size_t)N_NODES * N_HEADS * 4);
  float*    x1   = (float*)   carve((size_t)N_NODES * (N_HEADS * HID_F) * 4);
  float*    wh2  = (float*)   carve((size_t)N_NODES * OUT_F * 4);
  _Float16* wh2T = (_Float16*)carve((size_t)OUT_F * WHT_PITCH * 2);
  float*    s2s  = (float*)   carve((size_t)N_NODES * 4);
  float*    s2d  = (float*)   carve((size_t)N_NODES * 4);
  float*    m2   = (float*)   carve((size_t)N_NODES * 4);
  float*    l2   = (float*)   carve((size_t)N_NODES * 4);

  // One-time 100MB adjacency read -> 3.1MB L2-resident bitmask
  gat_pack_adj<<<dim3(N_NODES, (N_WORDS + 7) / 8), 256, 0, stream>>>(adj, bits);

  // ---- layer 1 ----
  gat_feat<IN_F, N_HEADS * HID_F>
      <<<(N_NODES * (N_HEADS * HID_F) + 255) / 256, 256, 0, stream>>>(hIn, W1, wh1, wh1T);
  gat_scores<N_HEADS, HID_F>
      <<<(N_NODES * N_HEADS + 255) / 256, 256, 0, stream>>>(wh1, a1s, a1d, s1s, s1d);
  gat_softmax_stats<N_HEADS>
      <<<(N_NODES + 7) / 8, 256, 0, stream>>>(bits, s1s, s1d, m1, l1);
  gat_aggregate<N_HEADS, HID_F, true>
      <<<(N_NODES + 15) / 16, 32, 0, stream>>>(bits, wh1T, s1s, s1d, m1, l1, x1);

  // ---- layer 2 ----
  gat_feat<N_HEADS * HID_F, OUT_F>
      <<<(N_NODES * OUT_F + 255) / 256, 256, 0, stream>>>(x1, W2, wh2, wh2T);
  gat_scores<1, OUT_F>
      <<<(N_NODES + 255) / 256, 256, 0, stream>>>(wh2, a2s, a2d, s2s, s2d);
  gat_softmax_stats<1>
      <<<(N_NODES + 7) / 8, 256, 0, stream>>>(bits, s2s, s2d, m2, l2);
  gat_aggregate<1, OUT_F, false>
      <<<(N_NODES + 15) / 16, 32, 0, stream>>>(bits, wh2T, s2s, s2d, m2, l2, out);
}
